// WARP_63599875719307
// MI455X (gfx1250) — matrix operand order repaired and checked by reading, not compiled, not run on hardware
//
#include <hip/hip_runtime.h>
#include <hip/hip_bf16.h>

// ---------------------------------------------------------------------------
// Problem constants
// ---------------------------------------------------------------------------
#define PREAL 9025            // P
#define KPAD  9216            // P padded to 288*32 -> 72 K-chunks per wave (even)
#define RPAD  9040            // P padded to 565*16 (WMMA M-tiles of 16)
#define NT    32              // L timesteps == GEMM N
#define KQ    (72 * 32)       // K-quarter per wave (4 waves x 72 chunks = 288)

typedef __attribute__((ext_vector_type(16))) __bf16 v16bf;
typedef __attribute__((ext_vector_type(8)))  __bf16 v8bf;
typedef __attribute__((ext_vector_type(8)))  float  v8f;
typedef __attribute__((ext_vector_type(4)))  float  v4f;

union U16bf { v16bf v; v8bf h[2]; };

// ---------------------------------------------------------------------------
// Kernel 1: A' = (A - I) -> bf16, zero-padded to RPAD x KPAD  (one-time pass)
// ---------------------------------------------------------------------------
__global__ __launch_bounds__(256) void prep_A_kernel(const float* __restrict__ A,
                                                     __bf16* __restrict__ Ap) {
  int c = blockIdx.x * 256 + threadIdx.x;
  int r = blockIdx.y;
  if (c >= KPAD) return;
  float v = 0.0f;
  if (r < PREAL && c < PREAL) {
    v = A[(size_t)r * PREAL + c];
    if (r == c) v -= 1.0f;
  }
  Ap[(size_t)r * KPAD + c] = (__bf16)v;
}

// ---------------------------------------------------------------------------
// Kernel 2: hypernet first two layers -> h1 (128) in workspace
// ---------------------------------------------------------------------------
__global__ __launch_bounds__(128) void hyper_kernel(
    const float* __restrict__ x,
    const float* __restrict__ hW0, const float* __restrict__ hb0,
    const float* __restrict__ hW1, const float* __restrict__ hb1,
    float* __restrict__ h1g) {
  __shared__ float x0[8];
  __shared__ float h0[128];
  int tid = threadIdx.x;
  if (tid < 8) x0[tid] = x[tid];
  __syncthreads();
  float s = hb0[tid];
  #pragma unroll
  for (int i = 0; i < 8; i++) s += hW0[tid * 8 + i] * x0[i];
  h0[tid] = fmaxf(s, 0.0f);
  __syncthreads();
  s = hb1[tid];
  #pragma unroll 8
  for (int i = 0; i < 128; i++) s += hW1[tid * 128 + i] * h0[i];
  h1g[tid] = fmaxf(s, 0.0f);
}

// ---------------------------------------------------------------------------
// Kernel 3: build M_0 = [theta0, c_1..c_31] (c_t = B @ dx_{t-1}), init Theta,
//           zero K-padding region of both double buffers.
// Layout: M / Theta stored transposed as (NT x KPAD): element (row p, col t)
//         at [t*KPAD + p]  -> contiguous-in-p, ideal for GEMM B operand loads.
// ---------------------------------------------------------------------------
__global__ __launch_bounds__(256) void init_state_kernel(
    const float* __restrict__ x, const float* __restrict__ Bm,
    const float* __restrict__ h1g,
    const float* __restrict__ hW2, const float* __restrict__ hb2,
    float* __restrict__ Mf0, __bf16* __restrict__ Mb0,
    float* __restrict__ Mf1, __bf16* __restrict__ Mb1,
    float* __restrict__ Theta) {
  __shared__ float xs[NT * 8];   // 256 floats
  __shared__ float h1[128];
  int tid = threadIdx.x;
  if (tid < 128) h1[tid] = h1g[tid];
  xs[tid] = x[tid];
  __syncthreads();

  int r = blockIdx.x * 256 + tid;
  if (r >= KPAD) return;

  if (r < PREAL) {
    // theta0[r] = hW2[r,:] @ h1 + hb2[r]
    float t0 = hb2[r];
    const float* w = hW2 + (size_t)r * 128;
    #pragma unroll 8
    for (int i = 0; i < 128; i++) t0 += w[i] * h1[i];
    Mf0[r] = t0; Mb0[r] = (__bf16)t0; Theta[r] = t0;

    float b[8];
    #pragma unroll
    for (int i = 0; i < 8; i++) b[i] = Bm[(size_t)r * 8 + i];
    for (int t = 1; t < NT; t++) {
      float c = 0.0f;
      if (t >= 2) {               // dx[0] == 0 -> c_1 = 0
        #pragma unroll
        for (int i = 0; i < 8; i++)
          c += b[i] * (xs[(t - 1) * 8 + i] - xs[(t - 2) * 8 + i]);
      }
      size_t o = (size_t)t * KPAD + r;
      Mf0[o] = c; Mb0[o] = (__bf16)c; Theta[o] = c;
    }
  } else {
    // K-padding rows: keep both double buffers and Theta clean
    for (int t = 0; t < NT; t++) {
      size_t o = (size_t)t * KPAD + r;
      Mf0[o] = 0.0f; Mb0[o] = (__bf16)0.0f;
      Mf1[o] = 0.0f; Mb1[o] = (__bf16)0.0f;
      Theta[o] = 0.0f;
    }
  }
}

// ---------------------------------------------------------------------------
// Kernel 4 (the hot loop, launched 31x): M_k = M_{k-1} + A' @ M_{k-1}
//   - one block per 16-row tile (565 blocks), 4 waves split K in quarters
//   - v_wmma_f32_16x16x32_bf16, fp32 accumulate, LDS cross-wave reduce
//   - scalar (SGPR) loop bounds via readfirstlane; even trip count -> unroll 2
//   - epilogue: write fp32 + bf16 copies of M_k, accumulate Theta[t+k] += M_k[:,t]
// ---------------------------------------------------------------------------
__global__ __launch_bounds__(128) void scan_gemm_kernel(
    const __bf16* __restrict__ Ap,
    const __bf16* __restrict__ MbIn, const float* __restrict__ MfIn,
    __bf16* __restrict__ MbOut, float* __restrict__ MfOut,
    float* __restrict__ Theta, int kshift) {
  __shared__ float red[3 * 512];

  const int wave = __builtin_amdgcn_readfirstlane(threadIdx.x >> 5); // uniform
  const int lane = threadIdx.x & 31;
  const int m = lane & 15;        // A: row-in-tile, B/D: column (N)
  const int h = lane >> 4;        // lane half
  const int rowBase = blockIdx.x * 16;

  // A operand (16x32 bf16): lane-half h holds K = {8h..8h+7} U {16+8h..16+8h+7}
  const __bf16* aPtr  = Ap + (size_t)(rowBase + m) * KPAD + 8 * h;
  // B operand (32x16 bf16): lane-half h holds K = {8h..8h+7} U {16+8h..16+8h+7}
  const __bf16* b0Ptr = MbIn + (size_t)m * KPAD + 8 * h;         // N = 0..15
  const __bf16* b1Ptr = MbIn + (size_t)(m + 16) * KPAD + 8 * h;  // N = 16..31

  v8f acc0 = {};
  v8f acc1 = {};

  const int kBeg = wave * KQ;          // SGPR-resident
  const int kEnd = kBeg + KQ;          // == KPAD for wave 3, no clamp needed

  #pragma unroll 2
  for (int kc = kBeg; kc < kEnd; kc += 32) {
    U16bf a, b0, b1;
    a.h[0]  = *(const v8bf*)(aPtr + kc);
    a.h[1]  = *(const v8bf*)(aPtr + kc + 16);
    b0.h[0] = *(const v8bf*)(b0Ptr + kc);
    b0.h[1] = *(const v8bf*)(b0Ptr + kc + 16);
    b1.h[0] = *(const v8bf*)(b1Ptr + kc);
    b1.h[1] = *(const v8bf*)(b1Ptr + kc + 16);
    acc0 = __builtin_amdgcn_wmma_f32_16x16x32_bf16(false, a.v, false, b0.v,
                                                   (short)0, acc0, false, false);
    acc1 = __builtin_amdgcn_wmma_f32_16x16x32_bf16(false, a.v, false, b1.v,
                                                   (short)0, acc1, false, false);
  }

  // cross-wave K reduction via LDS
  if (wave > 0) {
    float* d = red + (wave - 1) * 512;
    #pragma unroll
    for (int v = 0; v < 8; v++) {
      d[v * 32 + lane]       = acc0[v];
      d[256 + v * 32 + lane] = acc1[v];
    }
  }
  __syncthreads();
  if (wave != 0) return;
  #pragma unroll
  for (int w = 0; w < 3; w++) {
    const float* d = red + w * 512;
    #pragma unroll
    for (int v = 0; v < 8; v++) {
      acc0[v] += d[v * 32 + lane];
      acc1[v] += d[256 + v * 32 + lane];
    }
  }

  // Epilogue. D layout: VGPR v, lane-half h -> global row rowBase + v + 8h,
  // column m (acc0) / m+16 (acc1). 8 rows per lane are contiguous in memory.
  const int rows = rowBase + 8 * h;
  #pragma unroll
  for (int sel = 0; sel < 2; sel++) {
    v8f acc = sel ? acc1 : acc0;
    int t = m + 16 * sel;
    size_t base = (size_t)t * KPAD + rows;

    v4f oldLo = *(const v4f*)(MfIn + base);
    v4f oldHi = *(const v4f*)(MfIn + base + 4);
    v4f nLo, nHi;
    #pragma unroll
    for (int v = 0; v < 4; v++) {
      nLo[v] = oldLo[v] + acc[v];      // M_k = (I + A') M_{k-1}
      nHi[v] = oldHi[v] + acc[4 + v];
    }
    *(v4f*)(MfOut + base)     = nLo;
    *(v4f*)(MfOut + base + 4) = nHi;

    v8bf nb;
    #pragma unroll
    for (int v = 0; v < 4; v++) {
      nb[v]     = (__bf16)nLo[v];
      nb[4 + v] = (__bf16)nHi[v];
    }
    *(v8bf*)(MbOut + base) = nb;

    int ts = t + kshift;               // theta_{t+k} += M_k[:, t]
    if (ts < NT) {
      float* tp = Theta + (size_t)ts * KPAD + rows;
      v4f t0 = *(const v4f*)tp;
      v4f t1 = *(const v4f*)(tp + 4);
      #pragma unroll
      for (int v = 0; v < 4; v++) { t0[v] += nLo[v]; t1[v] += nHi[v]; }
      *(v4f*)tp       = t0;
      *(v4f*)(tp + 4) = t1;
    }
  }
}

// ---------------------------------------------------------------------------
// Kernel 5: per-position root MLP (64x9 -> 64x64 -> 64x64 -> 1x64)
// theta offsets: W0@0 b0@576 W1@640 b1@4736 W2@4800 b2@8896 W3@8960 b3@9024
// ---------------------------------------------------------------------------
__global__ __launch_bounds__(64) void root_kernel(const float* __restrict__ Theta,
                                                  const float* __restrict__ x,
                                                  float* __restrict__ out) {
  const int l = blockIdx.x;
  const int j = threadIdx.x;
  const float* th = Theta + (size_t)l * KPAD;
  __shared__ float rin[9];
  __shared__ float hh[64];

  if (j < 8) rin[j] = x[l * 8 + j];
  if (j == 0) rin[8] = (float)l * (1.0f / 31.0f);   // tau = linspace(0,1,32)
  __syncthreads();

  // layer 0
  float s = th[576 + j];
  #pragma unroll
  for (int i = 0; i < 9; i++) s += th[j * 9 + i] * rin[i];
  hh[j] = fmaxf(s, 0.0f);
  __syncthreads();

  // layer 1
  s = th[4736 + j];
  {
    const float* w = th + 640 + j * 64;
    #pragma unroll 8
    for (int i = 0; i < 64; i++) s += w[i] * hh[i];
  }
  __syncthreads();
  hh[j] = fmaxf(s, 0.0f);
  __syncthreads();

  // layer 2
  s = th[8896 + j];
  {
    const float* w = th + 4800 + j * 64;
    #pragma unroll 8
    for (int i = 0; i < 64; i++) s += w[i] * hh[i];
  }
  __syncthreads();
  hh[j] = fmaxf(s, 0.0f);
  __syncthreads();

  // layer 3 (no relu)
  if (j == 0) {
    float o = th[9024];
    #pragma unroll 8
    for (int i = 0; i < 64; i++) o += th[8960 + i] * hh[i];
    out[l] = o;
  }
}

// ---------------------------------------------------------------------------
// Host launcher (graph-capture safe: only kernel launches on `stream`)
// ---------------------------------------------------------------------------
extern "C" void kernel_launch(void* const* d_in, const int* in_sizes, int n_in,
                              void* d_out, int out_size, void* d_ws, size_t ws_size,
                              hipStream_t stream) {
  (void)in_sizes; (void)n_in; (void)out_size; (void)ws_size;
  const float* x   = (const float*)d_in[0];
  const float* A   = (const float*)d_in[1];
  const float* Bm  = (const float*)d_in[2];
  const float* hW0 = (const float*)d_in[3];
  const float* hb0 = (const float*)d_in[4];
  const float* hW1 = (const float*)d_in[5];
  const float* hb1 = (const float*)d_in[6];
  const float* hW2 = (const float*)d_in[7];
  const float* hb2 = (const float*)d_in[8];
  float* out = (float*)d_out;

  // workspace carve-out (all 256B aligned)
  char* ws = (char*)d_ws;
  size_t off = 0;
  auto carve = [&](size_t bytes) -> void* {
    void* p = ws + off;
    off = (off + bytes + 255) & ~(size_t)255;
    return p;
  };
  __bf16* Ap    = (__bf16*)carve((size_t)RPAD * KPAD * sizeof(__bf16)); // 166.6 MB, L2-resident
  __bf16* Mb0   = (__bf16*)carve((size_t)NT * KPAD * sizeof(__bf16));
  __bf16* Mb1   = (__bf16*)carve((size_t)NT * KPAD * sizeof(__bf16));
  float*  Mf0   = (float*) carve((size_t)NT * KPAD * sizeof(float));
  float*  Mf1   = (float*) carve((size_t)NT * KPAD * sizeof(float));
  float*  Theta = (float*) carve((size_t)NT * KPAD * sizeof(float));
  float*  h1g   = (float*) carve(128 * sizeof(float));

  __bf16* Mb[2] = { Mb0, Mb1 };
  float*  Mf[2] = { Mf0, Mf1 };

  // one-time HBM pass: A' = (A - I) in bf16 (fits in 192 MB L2 thereafter)
  prep_A_kernel<<<dim3((KPAD + 255) / 256, RPAD), 256, 0, stream>>>(A, Ap);
  // hypernet -> h1
  hyper_kernel<<<1, 128, 0, stream>>>(x, hW0, hb0, hW1, hb1, h1g);
  // M_0 = [theta0, c_1..c_31], Theta init, padding zeroed
  init_state_kernel<<<(KPAD + 255) / 256, 256, 0, stream>>>(
      x, Bm, h1g, hW2, hb2, Mf[0], Mb[0], Mf[1], Mb[1], Theta);

  // 31 GEMM passes: M_k = A M_{k-1}; Theta[t+k] += M_k[:, t]
  for (int k = 1; k < NT; k++) {
    int src = (k - 1) & 1, dst = k & 1;
    scan_gemm_kernel<<<RPAD / 16, 128, 0, stream>>>(
        Ap, Mb[src], Mf[src], Mb[dst], Mf[dst], Theta, k);
  }

  // per-position root MLP -> out (32 floats)
  root_kernel<<<NT, 64, 0, stream>>>(Theta, x, out);
}
